// Decoder_38757784879455
// MI455X (gfx1250) — matrix operand order repaired and checked by reading, not compiled, not run on hardware
//
#include <hip/hip_runtime.h>
#include <hip/hip_bf16.h>

// Sizes from the reference
#define EE   64
#define HH   128
#define DD   256
#define MLP  512
#define BSEQ 32
#define PP   64
#define NN   (BSEQ*PP)

typedef __attribute__((ext_vector_type(16))) _Float16 v16h;
typedef __attribute__((ext_vector_type(8)))  float    v8f;

union HPack { _Float16 h[2]; unsigned u; };

// Workspace layout (float-index offsets into d_ws)
#define WS_C     0                    // 512 floats : c = b_s@Wm1_pe + b_v@Wm1_ve + bm1
#define WS_WSW   1024                 // 2x512      : W_s @ Wm1[0:64]
#define WS_WVW   2048                 // 2x512      : W_v @ Wm1[64:128]
#define WS_U     4096                 // N*512      : u_j
#define WS_VNEG  (4096 + NN*512)      // N*512      : -(pos_i,vel_i part)
#define WS_WM2P  (4096 + 2*NN*512)    // 65536 dwords: Wm2 packed f16 B-fragments

#define NEG_INF  (-__builtin_inff())
#define POS_INF  ( __builtin_inff())

// ---------------------------------------------------------------------------
// P1: tiny 2x64 @ 64x512 folds + bias fold.  1 block x 512 threads.
// ---------------------------------------------------------------------------
__global__ void prep_small(const float* __restrict__ Ws, const float* __restrict__ bs,
                           const float* __restrict__ Wv, const float* __restrict__ bv,
                           const float* __restrict__ Wm1, const float* __restrict__ bm1,
                           float* __restrict__ ws) {
  int k = threadIdx.x;              // 0..511
  float s0 = 0.f, s1 = 0.f, v0 = 0.f, v1 = 0.f, cc = 0.f;
  for (int e = 0; e < 64; ++e) {
    float m  = Wm1[e * 512 + k];          // pe rows
    s0 += Ws[e] * m;  s1 += Ws[64 + e] * m;  cc += bs[e] * m;
    float m2 = Wm1[(64 + e) * 512 + k];   // ve rows
    v0 += Wv[e] * m2; v1 += Wv[64 + e] * m2; cc += bv[e] * m2;
  }
  ws[WS_C + k]         = cc + bm1[k];
  ws[WS_WSW + k]       = s0;  ws[WS_WSW + 512 + k] = s1;
  ws[WS_WVW + k]       = v0;  ws[WS_WVW + 512 + k] = v1;
}

// ---------------------------------------------------------------------------
// P2: per-pedestrian u_j / v_i vectors (512 each).  N blocks x 256 threads.
// ---------------------------------------------------------------------------
__global__ void prep_uv(const float* __restrict__ pos, const float* __restrict__ vel,
                        const float* __restrict__ hst, const float* __restrict__ Wm1,
                        float* __restrict__ ws) {
  int p = blockIdx.x, tid = threadIdx.x;
  __shared__ float sh[HH];
  if (tid < HH) sh[tid] = hst[p * HH + tid];
  __syncthreads();
  float px = pos[p*2], py = pos[p*2+1], vx = vel[p*2], vy = vel[p*2+1];
  for (int half = 0; half < 2; ++half) {
    int k = tid + half * 256;
    float ua = px * ws[WS_WSW + k] + py * ws[WS_WSW + 512 + k]
             + vx * ws[WS_WVW + k] + vy * ws[WS_WVW + 512 + k];
    float hs = 0.f;
    for (int h = 0; h < HH; ++h) hs += sh[h] * Wm1[(128 + h) * 512 + k];
    ws[WS_U    + p * 512 + k] = ua + hs;
    ws[WS_VNEG + p * 512 + k] = -ua;
  }
}

// ---------------------------------------------------------------------------
// P3: pack Wm2 (512x256 f32) into f16 WMMA B-fragment order.
// B layout (16-bit, 32x16): lanes 0-15 hold K in {0-7,16-23} (2/VGPR over V0..7),
// lanes 16-31 hold K in {8-15,24-31}; lane&15 = N column.
// Fragment (kt,nt) stored as 256 consecutive dwords: [lane][vgpr].
// ---------------------------------------------------------------------------
__global__ void pack_wm2(const float* __restrict__ Wm2, float* __restrict__ ws) {
  unsigned* out = (unsigned*)(ws + WS_WM2P);
  int idx  = blockIdx.x * blockDim.x + threadIdx.x;   // 0..65535
  int v    = idx & 7;
  int lane = (idx >> 3) & 31;
  int tile = idx >> 8;                                // kt*16 + nt
  int kt = tile >> 4, nt = tile & 15;
  int n  = nt * 16 + (lane & 15);
  int k  = kt * 32 + ((lane >> 4) << 3) + 2 * (v & 3) + ((v >> 2) << 4);
  HPack hp;
  hp.h[0] = (_Float16)Wm2[k * 256 + n];
  hp.h[1] = (_Float16)Wm2[(k + 1) * 256 + n];
  out[idx] = hp.u;
}

// ---------------------------------------------------------------------------
// Main: one workgroup per (scene b, pedestrian i).  256 threads = 8 waves.
//   A: h1 = relu(u_j + (v_i + c))  (64x512), built in f16 A-fragment layout in LDS
//   GEMM:  h2pre = h1 @ Wm2  via v_wmma_f32_16x16x32_f16
//   Fused: +bm2, relu, branchless visibility-masked max/min pool over j
//   Tail:  out = relu([mx,mn] @ Wp + bp)
// ---------------------------------------------------------------------------
__global__ void __launch_bounds__(256)
social_pool(const float* __restrict__ ws, const float* __restrict__ pos,
            const float* __restrict__ before, const float* __restrict__ bm2,
            const float* __restrict__ Wp, const float* __restrict__ bp,
            float* __restrict__ out) {
  __shared__ unsigned ldsA[2 * 16 * 32 * 8];   // 32 rows x 512 K, f16 packed (32 KB)
  __shared__ float    ldsVC[512];              // v_i + c  (per-workgroup constant)
  __shared__ float    ldsMx[256], ldsMn[256];
  __shared__ int      ldsMask[64];
  __shared__ unsigned ldsMaskBits[2];
  __shared__ int      ldsCnt;

  const unsigned* Wm2p = (const unsigned*)(ws + WS_WM2P);
  const float* uvec = ws + WS_U;
  const float* vvec = ws + WS_VNEG;
  const float* cvec = ws + WS_C;

  int wg = blockIdx.x;
  int b = wg >> 6, i = wg & 63;
  int tid = threadIdx.x;
  int lane = tid & 31, wave = tid >> 5;

  // ---- Stage A: visibility mask (egg + 120-degree cone), faithful to ref ----
  float xi = pos[(b * 64 + i) * 2], yi = pos[(b * 64 + i) * 2 + 1];
  if (tid < 64) {
    float xb = before[(b * 64 + i) * 2], yb = before[(b * 64 + i) * 2 + 1];
    float xr = xi - xb, yr = yi - yb;
    float safe = (xr == 0.f) ? 1.f : fabsf(xr);
    float at = atanf(fabsf(yr) / safe);
    const float r90 = 1.5707963267948966f, r180 = 3.141592653589793f,
                r270 = 4.71238898038469f;
    float ang;
    if      (xr > 0.f && yr > 0.f) ang = at + r270;
    else if (xr < 0.f && yr > 0.f) ang = r90 - at;
    else if (xr < 0.f && yr < 0.f) ang = r90 + at;
    else if (xr > 0.f && yr < 0.f) ang = r270 - at;
    else if (xr == 0.f && yr > 0.f) ang = 0.f;
    else if (xr == 0.f && yr < 0.f) ang = r180;
    else if (yr == 0.f && xr > 0.f) ang = r270;
    else if (yr == 0.f && xr < 0.f) ang = r90;
    else ang = 0.f;
    float cd = cosf(ang), sd = sinf(ang);
    int j = tid;
    float dx = pos[(b * 64 + j) * 2] - xi, dy = pos[(b * 64 + j) * 2 + 1] - yi;
    float xt = cd * dx - sd * dy, yt = sd * dx + cd * dy;
    float res = (yt >= 0.f) ? (xt * xt + yt * yt * 0.25f) : (xt * xt + yt * yt);
    bool egg = res <= 1.f;
    const float bb = 3.4641016151377544f;        // sin(60deg) * (2/cos(60deg))
    float c1 = xt * 2.f + bb * yt;               // x*a - x1*y, x1 = -bb
    float c2 = xt * 2.f - bb * yt;               // x*a - x2*y, x2 = +bb
    bool cone = ((c1 > 0.f) && (c2 < 0.f)) || (c1 == 0.f) || (c2 == 0.f);
    bool m = egg && (yt >= 0.f) && cone && (j != i);
    ldsMask[j] = m ? 1 : 0;
  }
  // v_i + c, computed once per workgroup (512 values over 256 threads)
  {
    const float* vrow = vvec + (b * 64 + i) * 512;
    ldsVC[tid]       = vrow[tid]       + cvec[tid];
    ldsVC[tid + 256] = vrow[tid + 256] + cvec[tid + 256];
  }
  __syncthreads();
  if (tid == 0) {
    unsigned lo = 0, hi = 0;
    for (int j = 0; j < 32; ++j)  lo |= (unsigned)ldsMask[j] << j;
    for (int j = 0; j < 32; ++j)  hi |= (unsigned)ldsMask[j + 32] << j;
    ldsMaskBits[0] = lo; ldsMaskBits[1] = hi;
    ldsCnt = __popc(lo) + __popc(hi);
  }
  __syncthreads();
  unsigned mbitsLo = ldsMaskBits[0];
  unsigned mbitsHi = ldsMaskBits[1];

  int nt0 = wave * 2;                            // this wave owns N-tiles nt0, nt0+1
  float bm2n[2], mx[2], mn[2];
  bm2n[0] = bm2[nt0 * 16 + (lane & 15)];
  bm2n[1] = bm2[(nt0 + 1) * 16 + (lane & 15)];
  mx[0] = mx[1] = NEG_INF;
  mn[0] = mn[1] = POS_INF;

  int mrow = (lane >> 4) << 3;                   // C layout: row = v + 8*laneHi

  for (int jh = 0; jh < 2; ++jh) {               // two 32-row halves of j
    unsigned mbits = jh ? mbitsHi : mbitsLo;     // mask bits for rows of this half
    // ---- Stage B: h1 half, relu'd, f16-packed into A-fragment layout ----
    for (int it = tid; it < 32 * 256; it += 256) {
      int jloc = it >> 8, kp = it & 255, k = kp * 2;
      int j = jh * 32 + jloc;
      const float* urow = uvec + (b * 64 + j) * 512;
      float a0 = urow[k]     + ldsVC[k];
      float a1 = urow[k + 1] + ldsVC[k + 1];
      a0 = a0 > 0.f ? a0 : 0.f;
      a1 = a1 > 0.f ? a1 : 0.f;
      HPack hp; hp.h[0] = (_Float16)a0; hp.h[1] = (_Float16)a1;
      // A layout (16-bit 16x32): lanes0-15: K in {0-7,16-23}; lanes16-31: {8-15,24-31}
      int kt = k >> 5, kk = k & 31;
      int al = (jloc & 15) + (((kk >> 3) & 1) << 4);
      int av = ((kk >> 1) & 3) + ((kk & 16) ? 4 : 0);
      int tile = (jloc >> 4) * 16 + kt;
      ldsA[(tile * 32 + al) * 8 + av] = hp.u;
    }
    __syncthreads();

    // ---- Stage C: WMMA GEMM (32 x 256, K=512) + fused bias/relu/masked pool ----
    #pragma unroll
    for (int nti = 0; nti < 2; ++nti) {
      int nt = nt0 + nti;
      v8f acc0 = {};   // j-tile 0 of this half
      v8f acc1 = {};   // j-tile 1
      #pragma unroll 4
      for (int kt = 0; kt < 16; ++kt) {
        v16h bfr = *(const v16h*)(Wm2p + ((kt * 16 + nt) * 256 + lane * 8));
        v16h a0  = *(const v16h*)(&ldsA[((0 * 16 + kt) * 32 + lane) * 8]);
        v16h a1  = *(const v16h*)(&ldsA[((1 * 16 + kt) * 32 + lane) * 8]);
        acc0 = __builtin_amdgcn_wmma_f32_16x16x32_f16(false, a0, false, bfr,
                                                      (short)0, acc0, false, false);
        acc1 = __builtin_amdgcn_wmma_f32_16x16x32_f16(false, a1, false, bfr,
                                                      (short)0, acc1, false, false);
      }
      // Branchless masked pool: select +/-inf identities via v_cndmask,
      // no exec-mask manipulation.
      #pragma unroll
      for (int v = 0; v < 8; ++v) {
        int r0 = mrow + v;                       // row within half for j-tile 0
        bool b0 = (mbits >> r0) & 1u;
        float h2a = acc0[v] + bm2n[nti]; h2a = h2a > 0.f ? h2a : 0.f;
        mx[nti] = fmaxf(mx[nti], b0 ? h2a : NEG_INF);
        mn[nti] = fminf(mn[nti], b0 ? h2a : POS_INF);
        int r1 = 16 + mrow + v;                  // row within half for j-tile 1
        bool b1 = (mbits >> r1) & 1u;
        float h2b = acc1[v] + bm2n[nti]; h2b = h2b > 0.f ? h2b : 0.f;
        mx[nti] = fmaxf(mx[nti], b1 ? h2b : NEG_INF);
        mn[nti] = fminf(mn[nti], b1 ? h2b : POS_INF);
      }
    }
    __syncthreads();   // protect ldsA before next half overwrites it
  }

  // ---- combine lane halves (same column n lives in lane and lane^16) ----
  #pragma unroll
  for (int nti = 0; nti < 2; ++nti) {
    float omx = __shfl_xor(mx[nti], 16, 32);
    float omn = __shfl_xor(mn[nti], 16, 32);
    mx[nti] = fmaxf(mx[nti], omx);
    mn[nti] = fminf(mn[nti], omn);
    if (lane < 16) {
      ldsMx[(nt0 + nti) * 16 + lane] = mx[nti];
      ldsMn[(nt0 + nti) * 16 + lane] = mn[nti];
    }
  }
  __syncthreads();

  // ---- Stage D: pooled = relu([mx,mn] @ Wp + bp), one column per thread ----
  int n = tid;
  float s = bp[n];
  if (ldsCnt > 0) {
    for (int q = 0; q < 256; ++q) s += ldsMx[q] * Wp[q * 256 + n];
    for (int q = 0; q < 256; ++q) s += ldsMn[q] * Wp[(256 + q) * 256 + n];
  }
  out[(b * 64 + i) * 256 + n] = s > 0.f ? s : 0.f;
}

// ---------------------------------------------------------------------------
extern "C" void kernel_launch(void* const* d_in, const int* in_sizes, int n_in,
                              void* d_out, int out_size, void* d_ws, size_t ws_size,
                              hipStream_t stream) {
  (void)in_sizes; (void)n_in; (void)out_size; (void)ws_size;
  const float* hst    = (const float*)d_in[0];   // h_states (1,N,H)
  // d_in[1] = seq_start_end (uniform scenes; honored structurally)
  const float* pos    = (const float*)d_in[2];   // end_pos (N,2)
  const float* vel    = (const float*)d_in[3];   // end_velocity (N,2)
  const float* before = (const float*)d_in[4];   // before_end_pos (N,2)
  const float* Ws     = (const float*)d_in[5];
  const float* bs     = (const float*)d_in[6];
  const float* Wv     = (const float*)d_in[7];
  const float* bv     = (const float*)d_in[8];
  const float* Wm1    = (const float*)d_in[9];
  const float* bm1    = (const float*)d_in[10];
  const float* Wm2    = (const float*)d_in[11];
  const float* bm2    = (const float*)d_in[12];
  const float* Wp     = (const float*)d_in[13];
  const float* bp     = (const float*)d_in[14];
  float* ws  = (float*)d_ws;
  float* outp = (float*)d_out;

  prep_small<<<1, 512, 0, stream>>>(Ws, bs, Wv, bv, Wm1, bm1, ws);
  prep_uv<<<NN, 256, 0, stream>>>(pos, vel, hst, Wm1, ws);
  pack_wm2<<<256, 256, 0, stream>>>(Wm2, ws);
  social_pool<<<NN, 256, 0, stream>>>(ws, pos, before, bm2, Wp, bp, outp);
}
